// MultiHeadAttention_63256278335882
// MI455X (gfx1250) — matrix operand order repaired and checked by reading, not compiled, not run on hardware
//
#include <hip/hip_runtime.h>

// ---------------------------------------------------------------------------
// MHA forward for MI455X (gfx1250): f16 WMMA everywhere, fp32 accumulate.
// Attention stages K/V tiles into LDS with async global->LDS copies
// (ASYNCcnt-tracked), double-buffered against the WMMA pipeline.
// ---------------------------------------------------------------------------

#define D_MODEL 1024
#define N_HEADS 16
#define D_HEAD  64
#define BATCH   2
#define SEQ     2048
#define MROWS   (BATCH * SEQ)          // 4096 rows of X

typedef _Float16 half4v  __attribute__((ext_vector_type(4)));
typedef _Float16 half8   __attribute__((ext_vector_type(8)));
typedef _Float16 half16  __attribute__((ext_vector_type(16)));
typedef float    float4v __attribute__((ext_vector_type(4)));
typedef float    float8  __attribute__((ext_vector_type(8)));
typedef int      int4v   __attribute__((ext_vector_type(4)));

typedef __attribute__((address_space(1))) int4v* g_b128_ptr;
typedef __attribute__((address_space(3))) int4v* l_b128_ptr;

union ABFrag { half16 v; half8 h[2]; };

static __device__ inline float8 zero8() {
    float8 z;
#pragma unroll
    for (int i = 0; i < 8; ++i) z[i] = 0.0f;
    return z;
}

static __device__ inline float8 wmma_f16(half16 a, half16 b, float8 c) {
    // D = A(16x32) * B(32x16) + C, fp32 accumulate
    return __builtin_amdgcn_wmma_f32_16x16x32_f16(
        /*neg_a=*/false, a, /*neg_b=*/false, b,
        /*c_mod=*/(short)0, c, /*reuse_a=*/false, /*reuse_b=*/false);
}

// ---- async global->LDS b128 copy (ASYNCcnt-tracked) ------------------------
static __device__ __forceinline__ void async_copy_b128(const _Float16* g,
                                                       _Float16* l) {
#if __has_builtin(__builtin_amdgcn_global_load_async_to_lds_b128)
    __builtin_amdgcn_global_load_async_to_lds_b128(
        (g_b128_ptr)g, (l_b128_ptr)l, /*offset=*/0, /*cpol=*/0);
#else
    asm volatile("global_load_async_to_lds_b128 %0, %1, off"
                 :: "v"((unsigned)(size_t)l), "v"(g)
                 : "memory");
#endif
}

static __device__ __forceinline__ void wait_async_all() {
#if __has_builtin(__builtin_amdgcn_s_wait_asynccnt)
    __builtin_amdgcn_s_wait_asynccnt(0);
#else
    asm volatile("s_wait_asynccnt 0x0" ::: "memory");
#endif
}

// ---------------------------------------------------------------------------
// fp32 -> fp16 bulk convert (vectorized x4)
// ---------------------------------------------------------------------------
__global__ void cvt_f32_f16(const float* __restrict__ src,
                            _Float16* __restrict__ dst, int n4) {
    int i = blockIdx.x * blockDim.x + threadIdx.x;
    if (i >= n4) return;
    float4v x = ((const float4v*)src)[i];
    half4v y;
    y[0] = (_Float16)x[0]; y[1] = (_Float16)x[1];
    y[2] = (_Float16)x[2]; y[3] = (_Float16)x[3];
    ((half4v*)dst)[i] = y;
}

// ---------------------------------------------------------------------------
// C[M,N] = A[M,K] @ W[N,K]^T   (A, W are f16 row-major)
// One wave computes a 64(M) x 32(N) tile: 8 WMMAs per 32-wide K step.
// Output modes:
//   0: Q  -> f16 [B,H,S,64]   (scaled by oscale = 1/sqrt(Dh))
//   1: K  -> f16 [B,H,S,64]
//   2: V  -> f16 [B,H,64,S]   (transposed for the PV matmul)
//   3: O  -> f32 [M,N]        (final output)
// ---------------------------------------------------------------------------
__global__ __launch_bounds__(256)
void gemm_xwT(const _Float16* __restrict__ A, const _Float16* __restrict__ W,
              void* __restrict__ out, int M, int N, int K,
              int mode, float oscale) {
    const int lane = threadIdx.x & 31;
    const int wave = threadIdx.x >> 5;
    const int hi   = lane >> 4;        // which 16-lane half of the wave
    const int l15  = lane & 15;

    const int tilesN = N / 32;
    const int wtile  = blockIdx.x * 8 + wave;
    const int tm     = (wtile / tilesN) * 64;
    const int tn     = (wtile % tilesN) * 32;

    float8 acc[4][2];
#pragma unroll
    for (int i = 0; i < 4; ++i)
#pragma unroll
        for (int j = 0; j < 2; ++j) acc[i][j] = zero8();

    for (int kb = 0; kb < K; kb += 32) {
        ABFrag a[4], b[2];
#pragma unroll
        for (int i = 0; i < 4; ++i) {
            const _Float16* ap = A + (size_t)(tm + i * 16 + l15) * K + kb + hi * 8;
            a[i].h[0] = *(const half8*)ap;          // K = hi*8 .. +8
            a[i].h[1] = *(const half8*)(ap + 16);   // K = 16+hi*8 .. +8
        }
#pragma unroll
        for (int j = 0; j < 2; ++j) {
            const _Float16* bp = W + (size_t)(tn + j * 16 + l15) * K + kb + hi * 8;
            b[j].h[0] = *(const half8*)bp;          // K = hi*8 .. +8
            b[j].h[1] = *(const half8*)(bp + 16);   // K = 16+hi*8 .. +8
        }
#pragma unroll
        for (int i = 0; i < 4; ++i)
#pragma unroll
            for (int j = 0; j < 2; ++j)
                acc[i][j] = wmma_f16(a[i].v, b[j].v, acc[i][j]);
    }

    // Epilogue: C frag layout = VGPR g holds row (hi*8+g), col = l15
#pragma unroll
    for (int i = 0; i < 4; ++i) {
#pragma unroll
        for (int j = 0; j < 2; ++j) {
#pragma unroll
            for (int g = 0; g < 8; ++g) {
                const int m = tm + i * 16 + hi * 8 + g;
                const int c = tn + j * 16 + l15;
                const float val = acc[i][j][g] * oscale;
                if (mode == 3) {
                    ((float*)out)[(size_t)m * N + c] = val;
                } else {
                    const int bb = m >> 11;          // m / SEQ
                    const int s  = m & (SEQ - 1);
                    const int h  = c >> 6;           // c / D_HEAD
                    const int d  = c & 63;
                    size_t addr;
                    if (mode == 2)   // V transposed: [B,H,64,S]
                        addr = ((size_t)(bb * N_HEADS + h) * D_HEAD + d) * SEQ + s;
                    else             // Q/K: [B,H,S,64]
                        addr = ((size_t)(bb * N_HEADS + h) * SEQ + s) * D_HEAD + d;
                    ((_Float16*)out)[addr] = (_Float16)val;
                }
            }
        }
    }
}

// ---------------------------------------------------------------------------
// Causal flash attention, async-LDS staged.
// Block = 8 waves = 128 query rows of one (b,h). One wave owns 16 q rows.
// Per kv tile (32 wide): K tile 32x64 and V tile 64x32 are DMA'd to LDS once
// per block (async b128, double buffered), all waves read fragments from LDS.
// Q: [B,H,S,64] (pre-scaled by 1/8), K: [B,H,S,64], VT: [B,H,64,S].
// ctx out: f16 [B,S,1024] so the output projection is a plain GEMM.
// ---------------------------------------------------------------------------
__global__ __launch_bounds__(256)
void attn_kernel(const _Float16* __restrict__ Q, const _Float16* __restrict__ Kx,
                 const _Float16* __restrict__ VT, _Float16* __restrict__ ctx) {
    __shared__ __align__(16) _Float16 Ks[2][32][64];   // 8 KB  (kv, d)
    __shared__ __align__(16) _Float16 Vs[2][64][32];   // 8 KB  (d, kv)
    __shared__ __align__(16) _Float16 Ps[8][16][32];   // 8 KB  per-wave P

    const int tid  = threadIdx.x;
    const int lane = tid & 31;
    const int wave = tid >> 5;
    const int hi   = lane >> 4;
    const int l15  = lane & 15;

    const int bh   = blockIdx.x & 31;      // 32 (b,h) pairs
    const int qblk = blockIdx.x >> 5;      // 16 q-blocks of 128 rows
    const int b    = bh >> 4;
    const int h    = bh & 15;
    const int qb   = qblk * 128;
    const int q0   = qb + wave * 16;

    const _Float16* Qh = Q  + (size_t)bh * SEQ * D_HEAD;
    const _Float16* Kh = Kx + (size_t)bh * SEQ * D_HEAD;
    const _Float16* Vh = VT + (size_t)bh * D_HEAD * SEQ;

    // Q fragments for this wave's 16 rows (K = 64 -> 2 chunks of 32)
    ABFrag qa[2];
#pragma unroll
    for (int c = 0; c < 2; ++c) {
        const _Float16* qp = Qh + (size_t)(q0 + l15) * D_HEAD + c * 32 + hi * 8;
        qa[c].h[0] = *(const half8*)qp;
        qa[c].h[1] = *(const half8*)(qp + 16);
    }

    float8 mrow, lrow, o[4];
#pragma unroll
    for (int g = 0; g < 8; ++g) { mrow[g] = -1e30f; lrow[g] = 0.0f; }
#pragma unroll
    for (int j = 0; j < 4; ++j) o[j] = zero8();

    // per-block async stage of one kv tile into LDS buffer `buf`
    auto issue_tile = [&](int t, int buf) {
        const int kv0 = t * 32;
        // K tile: 32 rows x 64 halfs = 256 x 16B chunks; 1 chunk/thread
        async_copy_b128(Kh + (size_t)(kv0 + (tid >> 3)) * D_HEAD + (tid & 7) * 8,
                        &Ks[buf][tid >> 3][(tid & 7) * 8]);
        // V tile: 64 rows (d) x 32 halfs (kv) from VT (row stride SEQ)
        async_copy_b128(Vh + (size_t)(tid >> 2) * SEQ + kv0 + (tid & 3) * 8,
                        &Vs[buf][tid >> 2][(tid & 3) * 8]);
    };

    const int nt = (qb + 128) / 32;        // uniform trip count for the block
    issue_tile(0, 0);

    for (int t = 0; t < nt; ++t) {
        wait_async_all();                  // own async ops complete
        __syncthreads();                   // tile t visible block-wide;
                                           // everyone done with buf[(t+1)&1]
        if (t + 1 < nt) issue_tile(t + 1, (t + 1) & 1);

        const int kv0 = t * 32;
        if (kv0 < q0 + 16) {               // causal bound for this wave
            const int buf = t & 1;

            // ---- scores: S(16x32) = Q(16x64) @ K_tile^T ---------------
            float8 s[2];
#pragma unroll
            for (int f = 0; f < 2; ++f) {
                ABFrag kb0, kb1;
                const _Float16* kp = &Ks[buf][f * 16 + l15][0];
                kb0.h[0] = *(const half8*)(kp + hi * 8);
                kb0.h[1] = *(const half8*)(kp + 16 + hi * 8);
                kb1.h[0] = *(const half8*)(kp + 32 + hi * 8);
                kb1.h[1] = *(const half8*)(kp + 48 + hi * 8);
                float8 z = zero8();
                z = wmma_f16(qa[0].v, kb0.v, z);
                z = wmma_f16(qa[1].v, kb1.v, z);
                s[f] = z;
            }

            // ---- causal mask (only tiles touching the diagonal) -------
            if (kv0 + 31 > q0) {
#pragma unroll
                for (int f = 0; f < 2; ++f)
#pragma unroll
                    for (int g = 0; g < 8; ++g) {
                        const int col = kv0 + f * 16 + l15;
                        const int row = q0 + hi * 8 + g;
                        if (col > row) s[f][g] = -1e30f;
                    }
            }

            // ---- online softmax --------------------------------------
            float8 tmx;
#pragma unroll
            for (int g = 0; g < 8; ++g) tmx[g] = fmaxf(s[0][g], s[1][g]);
#pragma unroll
            for (int off = 8; off >= 1; off >>= 1)
#pragma unroll
                for (int g = 0; g < 8; ++g)
                    tmx[g] = fmaxf(tmx[g], __shfl_xor(tmx[g], off, 32));

            float8 mnew, sc, p0, p1, rs;
#pragma unroll
            for (int g = 0; g < 8; ++g) {
                mnew[g] = fmaxf(mrow[g], tmx[g]);
                sc[g]   = __expf(mrow[g] - mnew[g]);
                p0[g]   = __expf(s[0][g] - mnew[g]);
                p1[g]   = __expf(s[1][g] - mnew[g]);
                rs[g]   = p0[g] + p1[g];
            }
#pragma unroll
            for (int off = 8; off >= 1; off >>= 1)
#pragma unroll
                for (int g = 0; g < 8; ++g)
                    rs[g] += __shfl_xor(rs[g], off, 32);
#pragma unroll
            for (int g = 0; g < 8; ++g) {
                lrow[g] = lrow[g] * sc[g] + rs[g];
                mrow[g] = mnew[g];
            }
#pragma unroll
            for (int j = 0; j < 4; ++j)
#pragma unroll
                for (int g = 0; g < 8; ++g) o[j][g] *= sc[g];

            // ---- restripe P (C-layout -> A-layout) via LDS ------------
#pragma unroll
            for (int g = 0; g < 8; ++g) {
                Ps[wave][hi * 8 + g][l15]      = (_Float16)p0[g];
                Ps[wave][hi * 8 + g][16 + l15] = (_Float16)p1[g];
            }
            ABFrag pa;   // same-wave LDS ops are in-order
            pa.h[0] = *(const half8*)&Ps[wave][l15][hi * 8];
            pa.h[1] = *(const half8*)&Ps[wave][l15][16 + hi * 8];

            // ---- O += P(16x32) @ V_tile(32x64) ------------------------
#pragma unroll
            for (int j = 0; j < 4; ++j) {
                ABFrag vb;
                const _Float16* vp = &Vs[buf][j * 16 + l15][hi * 8];
                vb.h[0] = *(const half8*)vp;
                vb.h[1] = *(const half8*)(vp + 16);
                o[j] = wmma_f16(pa.v, vb.v, o[j]);
            }
        }
    }

    // ---- normalize and store ctx as [B,S,D_MODEL] f16 -----------------
#pragma unroll
    for (int g = 0; g < 8; ++g) {
        const float inv = 1.0f / lrow[g];
        const int row = q0 + hi * 8 + g;
#pragma unroll
        for (int j = 0; j < 4; ++j) {
            const size_t addr = ((size_t)(b * SEQ + row) * D_MODEL)
                              + h * D_HEAD + j * 16 + l15;
            ctx[addr] = (_Float16)(o[j][g] * inv);
        }
    }
}

// ---------------------------------------------------------------------------
extern "C" void kernel_launch(void* const* d_in, const int* in_sizes, int n_in,
                              void* d_out, int out_size, void* d_ws, size_t ws_size,
                              hipStream_t stream) {
    const float* q  = (const float*)d_in[0];
    const float* k  = (const float*)d_in[1];
    const float* v  = (const float*)d_in[2];
    // d_in[3]: causal mask -- applied analytically in attn_kernel
    const float* Wq = (const float*)d_in[4];
    const float* Wk = (const float*)d_in[5];
    const float* Wv = (const float*)d_in[6];
    const float* Wo = (const float*)d_in[7];

    const size_t XE = (size_t)MROWS * D_MODEL;    // 4M elems
    const size_t WE = (size_t)D_MODEL * D_MODEL;  // 1M elems

    _Float16* p     = (_Float16*)d_ws;
    _Float16* Xq16  = p; p += XE;
    _Float16* Xk16  = p; p += XE;
    _Float16* Xv16  = p; p += XE;
    _Float16* Wq16  = p; p += WE;
    _Float16* Wk16  = p; p += WE;
    _Float16* Wv16  = p; p += WE;
    _Float16* Wo16  = p; p += WE;
    _Float16* Q16   = p; p += XE;   // [B,H,S,64]
    _Float16* K16   = p; p += XE;   // [B,H,S,64]
    _Float16* VT16  = p; p += XE;   // [B,H,64,S]
    _Float16* ctx16 = p; p += XE;   // [B,S,1024]

    const int n4x = (int)(XE / 4);
    const int n4w = (int)(WE / 4);
    cvt_f32_f16<<<(n4x + 255) / 256, 256, 0, stream>>>(q,  Xq16, n4x);
    cvt_f32_f16<<<(n4x + 255) / 256, 256, 0, stream>>>(k,  Xk16, n4x);
    cvt_f32_f16<<<(n4x + 255) / 256, 256, 0, stream>>>(v,  Xv16, n4x);
    cvt_f32_f16<<<(n4w + 255) / 256, 256, 0, stream>>>(Wq, Wq16, n4w);
    cvt_f32_f16<<<(n4w + 255) / 256, 256, 0, stream>>>(Wk, Wk16, n4w);
    cvt_f32_f16<<<(n4w + 255) / 256, 256, 0, stream>>>(Wv, Wv16, n4w);
    cvt_f32_f16<<<(n4w + 255) / 256, 256, 0, stream>>>(Wo, Wo16, n4w);

    // 64x32 tiles: (4096/64)*(1024/32) = 2048 waves -> 256 blocks of 8 waves
    const int gemmBlocks = (MROWS / 64) * (D_MODEL / 32) / 8;
    gemm_xwT<<<gemmBlocks, 256, 0, stream>>>(Xq16, Wq16, Q16,  MROWS, D_MODEL, D_MODEL, 0, 0.125f);
    gemm_xwT<<<gemmBlocks, 256, 0, stream>>>(Xk16, Wk16, K16,  MROWS, D_MODEL, D_MODEL, 1, 1.0f);
    gemm_xwT<<<gemmBlocks, 256, 0, stream>>>(Xv16, Wv16, VT16, MROWS, D_MODEL, D_MODEL, 2, 1.0f);

    // 32 (b,h) * 16 q-blocks of 128 rows = 512 blocks
    attn_kernel<<<512, 256, 0, stream>>>(Q16, K16, VT16, ctx16);

    gemm_xwT<<<gemmBlocks, 256, 0, stream>>>(ctx16, Wo16, d_out, MROWS, D_MODEL, D_MODEL, 3, 1.0f);
}
